// StateSpaceLayer_82094004895750
// MI455X (gfx1250) — matrix operand order repaired and checked by reading, not compiled, not run on hardware
//
#include <hip/hip_runtime.h>

typedef __attribute__((ext_vector_type(16))) _Float16 v16h;
typedef __attribute__((ext_vector_type(8)))  float    v8f;
typedef unsigned int v4u __attribute__((ext_vector_type(4)));
typedef int          v8i __attribute__((ext_vector_type(8)));
typedef int          v4i __attribute__((ext_vector_type(4)));

#define SEQLEN 1024
#define DDIM   64
#define CHUNK  64
#define NCH    (SEQLEN / CHUNK)
#define XS     68   // f32 LDS row stride for X tile (TDM pad: 64 DW + 4 DW)
#define MS     72   // f16 LDS row stride for M tile (bank-spread)
#define XBUFB  (CHUNK * XS * 4)   // bytes per X tile buffer

__device__ __forceinline__ float fast_exp(float x) {
  return __builtin_amdgcn_exp2f(x * 1.44269504088896340736f);   // v_exp_f32
}

__device__ __forceinline__ float fast_softplus(float x) {
  float m = fmaxf(x, 0.0f);
  float e = __builtin_amdgcn_exp2f(-fabsf(x) * 1.44269504088896340736f);
  return m + __builtin_amdgcn_logf(1.0f + e) * 0.69314718055994530942f;  // v_log_f32
}

// Tensor Data Mover: DMA one 64x64 f32 tile (row stride 4096 elems) into LDS,
// padding 4 DWORDs after every 64 DWORDs -> LDS row stride of XS=68 floats.
__device__ __forceinline__ void tdm_load_tile(const float* gptr, unsigned lds_off) {
  unsigned long long ga = (unsigned long long)(uintptr_t)gptr;
  v4u g0;
  g0.x = 1u;                                                  // count=1 (valid user D#)
  g0.y = lds_off;                                             // lds_addr (bytes)
  g0.z = (unsigned)(ga & 0xffffffffu);                        // global_addr[31:0]
  g0.w = (unsigned)((ga >> 32) & 0x01ffffffu) | (2u << 30);   // addr[56:32] | type=2
  v8i g1;
  g1[0] = (int)0x07520000;   // data_size=4B | pad_enable | pad_interval=64DW | pad_amount=4DW
  g1[1] = (int)(64u << 16);  // tensor_dim0 = 64 (low16)
  g1[2] = (int)(64u << 16);  // tensor_dim0 hi=0 | tensor_dim1 = 64 (low16)
  g1[3] = (int)(64u << 16);  // tensor_dim1 hi=0 | tile_dim0 = 64
  g1[4] = (int)64;           // tile_dim1 = 64 | tile_dim2 = 0
  g1[5] = (int)(DDIM * DDIM);// tensor_dim0_stride = 4096 elements (low32)
  g1[6] = 0;                 // stride0 hi | stride1 lo
  g1[7] = 0;                 // stride1 hi
  v4i gz = {0, 0, 0, 0};
#if __has_include(<hip/amd_detail/amd_gfx1250_TDM.h>)
  v8i gz8 = {0, 0, 0, 0, 0, 0, 0, 0};
  __builtin_amdgcn_tensor_load_to_lds(g0, g1, gz, gz, gz8, 0);  // clang-23 form
#else
  __builtin_amdgcn_tensor_load_to_lds(g0, g1, gz, gz, 0);       // ROCm 7.2 form
#endif
}

__global__ __launch_bounds__(256)
void ssm_chunked_wmma(const float* __restrict__ x,
                      const float* __restrict__ log_A,
                      const float* __restrict__ w_dt,
                      const float* __restrict__ b_dt,
                      float* __restrict__ out) {
  __shared__ float    Xlds[2][CHUNK * XS]; // double-buffered x tile, f32 (TDM target)
  __shared__ _Float16 Mlds[CHUNK * MS];    // causal decay matrix, f16
  __shared__ float    cslds[CHUNK];        // local inclusive cumsum of log_A_dt
  __shared__ float    carrylds[DDIM];      // cross-chunk state (= out row at chunk end)
  __shared__ float    wlds[DDIM];          // dt projection weights

  const int tid  = threadIdx.x;
  const int lane = tid & 31;               // wave32
  const int wave = tid >> 5;               // 8 waves
  const int n    = lane & 15;              // B/D column within tile
  const int hi   = lane >> 4;              // half-wave selector
  const int tr   = wave >> 1;              // 16-row strip (0..3)
  const int tc0  = (wave & 1) << 1;        // first of two 16-col tiles

  const int r = blockIdx.x & (DDIM - 1);
  const int b = blockIdx.x >> 6;

  const float aR  = -fast_exp(log_A[r]);   // A_continuous[r] < 0
  const float bdt = b_dt[0];

  // uniform (SGPR) leader predicate -> scalar branch, TDM issued by wave 0 only
  const bool leader = (__builtin_amdgcn_readfirstlane(tid) < 32);
  const unsigned xlds_base = (unsigned)(uintptr_t)(&Xlds[0][0]);  // LDS byte offset

  if (tid < DDIM) { wlds[tid] = w_dt[tid]; carrylds[tid] = 0.0f; }

  const size_t bbase = (size_t)b * SEQLEN * DDIM * DDIM + (size_t)r * DDIM;

  // prologue: start DMA of chunk 0 into buffer 0
  if (leader) tdm_load_tile(x + bbase, xlds_base);
  __syncthreads();

  for (int ch = 0; ch < NCH; ++ch) {
    const size_t cbase = bbase + (size_t)ch * CHUNK * (DDIM * DDIM);
    const float* Xb = &Xlds[ch & 1][0];

    // ---- 1. wait for this chunk's tile; kick off next chunk's DMA ----
    if (leader) {
      __builtin_amdgcn_s_wait_tensorcnt(0);
      if (ch + 1 < NCH)
        tdm_load_tile(x + cbase + (size_t)CHUNK * (DDIM * DDIM),
                      xlds_base + (unsigned)(((ch + 1) & 1) * XBUFB));
    }
    __syncthreads();

    // ---- 2. dt = softplus(x_row . w + b);  log_A_dt = dt * aR ----
    if (tid < CHUNK) {
      float dot = bdt;
      #pragma unroll 8
      for (int c = 0; c < DDIM; ++c) dot += Xb[tid * XS + c] * wlds[c];
      cslds[tid] = fast_softplus(dot) * aR;
    }
    __syncthreads();

    // ---- 3. inclusive Hillis-Steele scan over the 64 chunk rows ----
    for (int off = 1; off < CHUNK; off <<= 1) {
      float add = 0.0f;
      if (tid < CHUNK && tid >= off) add = cslds[tid - off];
      __syncthreads();
      if (tid < CHUNK) cslds[tid] += add;
      __syncthreads();
    }

    // ---- 4. build causal decay matrix M[t][j] = exp(cs[t]-cs[j]) (<=1), f16 ----
    {
      const int t  = tid >> 2;
      const int j0 = (tid & 3) << 4;
      const float cst = cslds[t];
      #pragma unroll
      for (int jj = 0; jj < 16; ++jj) {
        const int j = j0 + jj;
        const float m = (t >= j) ? fast_exp(cst - cslds[j]) : 0.0f;
        Mlds[t * MS + j] = (_Float16)m;
      }
    }
    __syncthreads();

    // ---- 5. per-wave WMMA: strip tr, two 16x16 output tiles, K=64 in 2 steps ----
    union U16 { v16h v; _Float16 h[16]; };
    union U8  { v8f  v; float    f[8];  };
    U16 a0, a1, b00, b01, b10, b11;

    const int mrow = tr * 16 + n;
    #pragma unroll
    for (int i = 0; i < 16; ++i) {
      // 16-bit A layout: VGPR0-3 hold K = hi*8 + 0..7, VGPR4-7 hold K = 16 + hi*8 + 0..7
      const int kk = (i < 8) ? (hi * 8 + i) : (16 + hi * 8 + (i - 8));
      a0.h[i] = Mlds[mrow * MS + kk];
      a1.h[i] = Mlds[mrow * MS + 32 + kk];
    }
    #pragma unroll
    for (int i = 0; i < 16; ++i) {
      // 16-bit B layout: lane holds column n, K = hi*8 + i (i<8), 16 + hi*8 + (i-8) (i>=8)
      const int kk = (i < 8) ? (hi * 8 + i) : (16 + hi * 8 + (i - 8));
      b00.h[i] = (_Float16)Xb[kk        * XS + (tc0 * 16 + n)];
      b10.h[i] = (_Float16)Xb[(32 + kk) * XS + (tc0 * 16 + n)];
      b01.h[i] = (_Float16)Xb[kk        * XS + ((tc0 + 1) * 16 + n)];
      b11.h[i] = (_Float16)Xb[(32 + kk) * XS + ((tc0 + 1) * 16 + n)];
    }

    // accumulator seeded with inter-chunk term: exp(cs_local[t]) * carry[c]
    U8 acc0, acc1;
    #pragma unroll
    for (int v = 0; v < 8; ++v) {
      const int tloc = tr * 16 + v + 8 * hi;   // C/D layout: VGPR v -> row v + 8*hi
      const float decay = fast_exp(cslds[tloc]);
      acc0.f[v] = decay * carrylds[tc0 * 16 + n];
      acc1.f[v] = decay * carrylds[(tc0 + 1) * 16 + n];
    }
    __syncthreads();   // all carry/cs reads done before carry is overwritten below

    acc0.v = __builtin_amdgcn_wmma_f32_16x16x32_f16(false, a0.v, false, b00.v, (short)0, acc0.v, false, false);
    acc0.v = __builtin_amdgcn_wmma_f32_16x16x32_f16(false, a1.v, false, b10.v, (short)0, acc0.v, false, false);
    acc1.v = __builtin_amdgcn_wmma_f32_16x16x32_f16(false, a0.v, false, b01.v, (short)0, acc1.v, false, false);
    acc1.v = __builtin_amdgcn_wmma_f32_16x16x32_f16(false, a1.v, false, b11.v, (short)0, acc1.v, false, false);

    // ---- 6. store output tiles (f32) and update carry with chunk's last row ----
    #pragma unroll
    for (int v = 0; v < 8; ++v) {
      const size_t orow = cbase + (size_t)(tr * 16 + v + 8 * hi) * (DDIM * DDIM);
      out[orow + tc0 * 16 + n]       = acc0.f[v];
      out[orow + (tc0 + 1) * 16 + n] = acc1.f[v];
    }
    if (tr == 3 && hi == 1) {   // local row 63 lives in VGPR7 of lanes 16..31, strip 3
      carrylds[tc0 * 16 + n]       = acc0.f[7];
      carrylds[(tc0 + 1) * 16 + n] = acc1.f[7];
    }
    __syncthreads();
  }
}

extern "C" void kernel_launch(void* const* d_in, const int* in_sizes, int n_in,
                              void* d_out, int out_size, void* d_ws, size_t ws_size,
                              hipStream_t stream) {
  const float* x     = (const float*)d_in[0];
  const float* log_A = (const float*)d_in[1];
  const float* w_dt  = (const float*)d_in[2];
  const float* b_dt  = (const float*)d_in[3];
  float* out = (float*)d_out;

  dim3 grid(2 * DDIM);   // B * D = 128 independent (batch, channel) scans
  dim3 block(256);       // 8 wave32s
  ssm_chunked_wmma<<<grid, block, 0, stream>>>(x, log_A, w_dt, b_dt, out);
}
